// Processor_4071628997409
// MI455X (gfx1250) — matrix operand order repaired and checked by reading, not compiled, not run on hardware
//
#include <hip/hip_runtime.h>

typedef __attribute__((ext_vector_type(16))) _Float16 v16h;
typedef __attribute__((ext_vector_type(8)))  _Float16 v8h;
typedef __attribute__((ext_vector_type(8)))  float    v8f;

#define NLAT 256
#define NPTS 8192
#define NVE  16
#define NB   2

// ---------------------------------------------------------------------------
// Fragment helpers (CDNA5 WMMA 16x16x32 f16, wave32)
//
// A (16x32, MxK):  row M = lane&15, element j holds K = k0 + (j>>3)*16 + hs*8 + (j&7),
//                  hs = lane>>4.   (ISA 7.12.2 "16-bit A-Matrix 16x32")
// B (32x16, KxN):  col N = lane&15, element j holds K = (j>>3)*16 + hs*8 + (j&7) (same as A).
// C/D (16x16):     col N = lane&15, VGPR v holds row M = v + 8*hs.
// ---------------------------------------------------------------------------

__device__ __forceinline__ v16h make_a16(v8h lo, v8h hi) {
  v16h a;
#pragma unroll
  for (int j = 0; j < 8; ++j) { a[j] = lo[j]; a[j + 8] = hi[j]; }
  return a;
}

// A-fragment from a global fp32 row (convert to f16 on the fly)
__device__ __forceinline__ v16h gmemA(const float* __restrict__ row, int k0, int hs) {
  const float4* p0 = reinterpret_cast<const float4*>(row + k0 + hs * 8);
  const float4* p1 = reinterpret_cast<const float4*>(row + k0 + 16 + hs * 8);
  float4 a0 = p0[0], a1 = p0[1];
  float4 b0 = p1[0], b1 = p1[1];
  v16h a;
  a[0] = (_Float16)a0.x; a[1] = (_Float16)a0.y; a[2] = (_Float16)a0.z; a[3] = (_Float16)a0.w;
  a[4] = (_Float16)a1.x; a[5] = (_Float16)a1.y; a[6] = (_Float16)a1.z; a[7] = (_Float16)a1.w;
  a[8]  = (_Float16)b0.x; a[9]  = (_Float16)b0.y; a[10] = (_Float16)b0.z; a[11] = (_Float16)b0.w;
  a[12] = (_Float16)b1.x; a[13] = (_Float16)b1.y; a[14] = (_Float16)b1.z; a[15] = (_Float16)b1.w;
  return a;
}

// A-fragment from an f16 row-major LDS activation buffer
__device__ __forceinline__ v16h ldsA(const _Float16* act, int row, int k0, int hs) {
  const v8h* p0 = reinterpret_cast<const v8h*>(act + row * NLAT + k0 + hs * 8);
  const v8h* p1 = reinterpret_cast<const v8h*>(act + row * NLAT + k0 + 16 + hs * 8);
  return make_a16(p0[0], p1[0]);
}

// B-fragment from pre-packed weights: fragment = 32 lanes * 16 halves contiguous (1KB)
__device__ __forceinline__ v16h loadB(const _Float16* __restrict__ wp, int frag, int lane) {
  const v8h* p = reinterpret_cast<const v8h*>(wp + (size_t)frag * 512 + lane * 16);
  return make_a16(p[0], p[1]);
}

__device__ __forceinline__ v8f wmma_f16(v16h a, v16h b, v8f c) {
  return __builtin_amdgcn_wmma_f32_16x16x32_f16(false, a, false, b, (short)0, c, false, false);
}

__device__ __forceinline__ void bias_init(v8f acc[16], const float* __restrict__ bias, int nlow) {
#pragma unroll
  for (int nt = 0; nt < 16; ++nt) {
    float bv = bias[nt * 16 + nlow];
    v8f z;
#pragma unroll
    for (int v = 0; v < 8; ++v) z[v] = bv;
    acc[nt] = z;
  }
}

__device__ __forceinline__ void store_act_relu(_Float16* act, int rowbase, int hs, int nlow,
                                               v8f acc[16]) {
#pragma unroll
  for (int nt = 0; nt < 16; ++nt) {
#pragma unroll
    for (int v = 0; v < 8; ++v) {
      float x = fmaxf(acc[nt][v], 0.0f);
      act[(rowbase + v + 8 * hs) * NLAT + nt * 16 + nlow] = (_Float16)x;
    }
  }
}

// one 16x256 row-tile times a 256x256 packed weight, accumulate into acc[16]
__device__ __forceinline__ void mm_from_lds(v8f acc[16], const _Float16* act, int rowA,
                                            int hs, int lane, const _Float16* __restrict__ w) {
  for (int c = 0; c < 8; ++c) {
    v16h a = ldsA(act, rowA, c * 32, hs);
#pragma unroll
    for (int nt = 0; nt < 16; ++nt) {
      v16h bf = loadB(w, c * 16 + nt, lane);
      acc[nt] = wmma_f16(a, bf, acc[nt]);
    }
  }
}

// ---------------------------------------------------------------------------
// Weight pack: fp32 [K,256] row-major -> f16 WMMA B-fragment order
// ---------------------------------------------------------------------------
__global__ void pack_w(const float* __restrict__ src, _Float16* __restrict__ dst, int total) {
  int tid = blockIdx.x * 256 + threadIdx.x;
  if (tid >= total) return;
  int f   = tid >> 9;          // fragment id = kt*16 + nt
  int rem = tid & 511;
  int l   = rem >> 4;          // lane 0..31
  int j   = rem & 15;          // element 0..15
  int nt  = f & 15;
  int kt  = f >> 4;
  int k   = kt * 32 + ((j >> 3) << 4) + ((l >> 4) << 3) + (j & 7);
  int n   = (nt << 4) + (l & 15);
  dst[tid] = (_Float16)src[k * NLAT + n];
}

// ---------------------------------------------------------------------------
// Edge MLP: [B,NP,NV] rows of [self|nb|edge] (768) -> 256 (relu) -> 256 (relu)
//           -> 256 (relu) -> 256; writes new_edges (b==0) and sum over NV.
// grid = 2048 blocks (b * 1024 + point-group), block = 256 thr = 8 waves,
// wave w handles point p = group*8 + w (16 edge rows = one M-tile).
// ---------------------------------------------------------------------------
__global__ void __launch_bounds__(256)
edge_mlp_kernel(const float* __restrict__ nodes, const float* __restrict__ eattr,
                const int* __restrict__ eidx,
                const _Float16* __restrict__ w1,  const float* __restrict__ b1,
                const _Float16* __restrict__ wh0, const float* __restrict__ bh0,
                const _Float16* __restrict__ wh1, const float* __restrict__ bh1,
                const _Float16* __restrict__ wl,  const float* __restrict__ bl,
                float* __restrict__ agg_ws, float* __restrict__ out_edges) {
  extern __shared__ _Float16 act[];                 // 128 rows x 256 f16 = 64 KB
  const int lane = threadIdx.x & 31;
  const int wave = threadIdx.x >> 5;
  const int hs   = lane >> 4;
  const int nlow = lane & 15;
  const int b    = blockIdx.x >> 10;
  const int p    = ((blockIdx.x & 1023) << 3) + wave;
  const int rowbase = wave << 4;
  const int rowA    = rowbase + nlow;

  const float* selfRow = nodes + ((size_t)b * NPTS + p) * NLAT;
  const int    nidx    = eidx[p * NVE + nlow];
  const float* nbRow   = nodes + ((size_t)b * NPTS + nidx) * NLAT;
  const float* edRow   = eattr + ((size_t)p * NVE + nlow) * NLAT;

  v8f acc[16];

  // ---- layer 1: K = 768, A sources = self / gathered-neighbor / edge-attr ----
  bias_init(acc, b1, nlow);
  for (int c = 0; c < 24; ++c) {
    const float* rp = (c < 8) ? selfRow : ((c < 16) ? nbRow : edRow);
    v16h a = gmemA(rp, (c & 7) * 32, hs);
#pragma unroll
    for (int nt = 0; nt < 16; ++nt) {
      v16h bf = loadB(w1, c * 16 + nt, lane);
      acc[nt] = wmma_f16(a, bf, acc[nt]);
    }
  }
  store_act_relu(act, rowbase, hs, nlow, acc);

  // ---- hidden layers (in-place LDS is safe: wave reads all K before writing) ----
  bias_init(acc, bh0, nlow);
  mm_from_lds(acc, act, rowA, hs, lane, wh0);
  store_act_relu(act, rowbase, hs, nlow, acc);

  bias_init(acc, bh1, nlow);
  mm_from_lds(acc, act, rowA, hs, lane, wh1);
  store_act_relu(act, rowbase, hs, nlow, acc);

  // ---- last layer (no relu) ----
  bias_init(acc, bl, nlow);
  mm_from_lds(acc, act, rowA, hs, lane, wl);

  // ---- epilogue: new_edges (b==0) and sum over the 16 neighbors (M dim) ----
#pragma unroll
  for (int nt = 0; nt < 16; ++nt) {
    float s = 0.0f;
#pragma unroll
    for (int v = 0; v < 8; ++v) {
      float x = acc[nt][v];
      s += x;
      if (b == 0)
        out_edges[((size_t)p * NVE + v + 8 * hs) * NLAT + nt * 16 + nlow] = x;
    }
    s += __shfl_xor(s, 16, 32);
    if (lane < 16)
      agg_ws[((size_t)b * NPTS + p) * NLAT + nt * 16 + lane] = s;
  }
}

// ---------------------------------------------------------------------------
// Node MLP: [B*NP] rows of [nodes|agg] (512) -> 256x3 relu -> 256, + residual.
// grid = 128 blocks, 8 waves each, wave handles 16 consecutive rows.
// ---------------------------------------------------------------------------
__global__ void __launch_bounds__(256)
node_mlp_kernel(const float* __restrict__ nodes, const float* __restrict__ agg_ws,
                const _Float16* __restrict__ w1,  const float* __restrict__ b1,
                const _Float16* __restrict__ wh0, const float* __restrict__ bh0,
                const _Float16* __restrict__ wh1, const float* __restrict__ bh1,
                const _Float16* __restrict__ wl,  const float* __restrict__ bl,
                float* __restrict__ out_nodes) {
  extern __shared__ _Float16 act[];
  const int lane = threadIdx.x & 31;
  const int wave = threadIdx.x >> 5;
  const int hs   = lane >> 4;
  const int nlow = lane & 15;
  const int r0   = (blockIdx.x * 8 + wave) * 16;   // global row base
  const int rowbase = wave << 4;
  const int rowA    = rowbase + nlow;

  const float* xRow = nodes  + (size_t)(r0 + nlow) * NLAT;
  const float* aRow = agg_ws + (size_t)(r0 + nlow) * NLAT;

  v8f acc[16];

  // ---- layer 1: K = 512, A sources = nodes / aggregated messages ----
  bias_init(acc, b1, nlow);
  for (int c = 0; c < 16; ++c) {
    const float* rp = (c < 8) ? xRow : aRow;
    v16h a = gmemA(rp, (c & 7) * 32, hs);
#pragma unroll
    for (int nt = 0; nt < 16; ++nt) {
      v16h bf = loadB(w1, c * 16 + nt, lane);
      acc[nt] = wmma_f16(a, bf, acc[nt]);
    }
  }
  store_act_relu(act, rowbase, hs, nlow, acc);

  bias_init(acc, bh0, nlow);
  mm_from_lds(acc, act, rowA, hs, lane, wh0);
  store_act_relu(act, rowbase, hs, nlow, acc);

  bias_init(acc, bh1, nlow);
  mm_from_lds(acc, act, rowA, hs, lane, wh1);
  store_act_relu(act, rowbase, hs, nlow, acc);

  bias_init(acc, bl, nlow);
  mm_from_lds(acc, act, rowA, hs, lane, wl);

  // ---- residual add + fp32 store ----
#pragma unroll
  for (int nt = 0; nt < 16; ++nt) {
#pragma unroll
    for (int v = 0; v < 8; ++v) {
      size_t oi = (size_t)(r0 + v + 8 * hs) * NLAT + nt * 16 + nlow;
      out_nodes[oi] = nodes[oi] + acc[nt][v];
    }
  }
}

// ---------------------------------------------------------------------------
extern "C" void kernel_launch(void* const* d_in, const int* in_sizes, int n_in,
                              void* d_out, int out_size, void* d_ws, size_t ws_size,
                              hipStream_t stream) {
  const float* nodes = (const float*)d_in[0];
  const float* eattr = (const float*)d_in[1];
  const int*   eidx  = (const int*)d_in[2];
  const float* e_fw  = (const float*)d_in[3];
  const float* e_fb  = (const float*)d_in[4];
  const float* e_hw  = (const float*)d_in[5];
  const float* e_hb  = (const float*)d_in[6];
  const float* e_lw  = (const float*)d_in[7];
  const float* e_lb  = (const float*)d_in[8];
  const float* n_fw  = (const float*)d_in[9];
  const float* n_fb  = (const float*)d_in[10];
  const float* n_hw  = (const float*)d_in[11];
  const float* n_hb  = (const float*)d_in[12];
  const float* n_lw  = (const float*)d_in[13];
  const float* n_lb  = (const float*)d_in[14];

  // workspace layout: packed f16 weights, then fp32 agg buffer
  _Float16* wb     = (_Float16*)d_ws;
  _Float16* e_fw_p = wb;                    // 768*256
  _Float16* e_h0_p = e_fw_p + 196608;       // 256*256
  _Float16* e_h1_p = e_h0_p + 65536;
  _Float16* e_lw_p = e_h1_p + 65536;
  _Float16* n_fw_p = e_lw_p + 65536;        // 512*256
  _Float16* n_h0_p = n_fw_p + 131072;
  _Float16* n_h1_p = n_h0_p + 65536;
  _Float16* n_lw_p = n_h1_p + 65536;
  float* agg_ws = (float*)((char*)d_ws + 1441792);  // B*NP*256 fp32 = 16 MB

  auto pk = [&](const float* s, _Float16* dst, int K) {
    int total = K * NLAT;
    pack_w<<<total / 256, 256, 0, stream>>>(s, dst, total);
  };
  pk(e_fw, e_fw_p, 768);
  pk(e_hw,         e_h0_p, 256);
  pk(e_hw + 65536, e_h1_p, 256);
  pk(e_lw, e_lw_p, 256);
  pk(n_fw, n_fw_p, 512);
  pk(n_hw,         n_h0_p, 256);
  pk(n_hw + 65536, n_h1_p, 256);
  pk(n_lw, n_lw_p, 256);

  float* out_nodes = (float*)d_out;
  float* out_edges = out_nodes + (size_t)NB * NPTS * NLAT;

  edge_mlp_kernel<<<2048, 256, 128 * NLAT * sizeof(_Float16), stream>>>(
      nodes, eattr, eidx,
      e_fw_p, e_fb, e_h0_p, e_hb, e_h1_p, e_hb + 256, e_lw_p, e_lb,
      agg_ws, out_edges);

  node_mlp_kernel<<<128, 256, 128 * NLAT * sizeof(_Float16), stream>>>(
      nodes, agg_ws,
      n_fw_p, n_fb, n_h0_p, n_hb, n_h1_p, n_hb + 256, n_lw_p, n_lb,
      out_nodes);
}
